// VectorQuantizer_15487652069633
// MI455X (gfx1250) — matrix operand order repaired and checked by reading, not compiled, not run on hardware
//
#include <hip/hip_runtime.h>
#include <hip/hip_bf16.h>

// Problem constants (match reference)
#define K_CODES 1024
#define E_DIM   64
#define T_DIM   1024
#define B_DIM   64
#define N_VEC   (T_DIM * B_DIM)     // 65536 vectors
#define BETA    0.25f
#define N_TILES (K_CODES / 16)      // 64 column tiles

typedef __attribute__((ext_vector_type(16))) __bf16 v16bf;
typedef __attribute__((ext_vector_type(8)))  float  v8f;
typedef __attribute__((ext_vector_type(4)))  unsigned v4u;
typedef __attribute__((ext_vector_type(8)))  unsigned v8u;

// ---- bf16 helpers (RNE split: x ~= hi + lo, each bf16) ----
static __device__ __forceinline__ unsigned short f2bf_rne(float f) {
    unsigned int u = __float_as_uint(f);
    u += 0x7FFFu + ((u >> 16) & 1u);
    return (unsigned short)(u >> 16);
}
static __device__ __forceinline__ float bf2f(unsigned short h) {
    return __uint_as_float(((unsigned int)h) << 16);
}

// ------------------------------------------------------------------
// TDM: DMA one 16x64 bf16 tile (one codebook sub-block) into LDS with
// hardware row padding 128B -> 144B (32 dwords data + 4 dwords pad).
// D# group0/group1 built in scalar regs; 2D tensor -> 2-group form.
// ------------------------------------------------------------------
static __device__ __forceinline__ void tdm_stage_tile(const unsigned short* gsrc,
                                                      unsigned lds_off) {
    unsigned long long ga = (unsigned long long)(size_t)gsrc;
    v4u g0;
    g0[0] = 1u;                                   // count=1, user descriptor
    g0[1] = lds_off;                              // lds_addr (bytes)
    g0[2] = (unsigned)(ga & 0xffffffffu);         // global_addr[31:0]
    g0[3] = (unsigned)((ga >> 32) & 0x01ffffffu)  // global_addr[56:32]
            | (2u << 30);                         // type = 2 ("image")
    v8u g1;
    g1[0] = (1u << 16)       // data_size = 1 -> 2 bytes
          | (1u << 20)       // pad_enable
          | (4u << 22)       // pad_interval: 32 dwords (one 128B row)
          | (3u << 25);      // pad_amount: 4 dwords (16B)
    g1[1] = (64u << 16);     // tensor_dim0 = 64 elements (bits 79:48)
    g1[2] = (1024u << 16);   // tensor_dim1 = 1024 rows   (bits 111:80)
    g1[3] = (64u << 16);     // tile_dim0 = 64            (bits 127:112)
    g1[4] = 16u;             // tile_dim1 = 16 rows       (bits 143:128)
    g1[5] = 64u;             // tensor_dim0_stride = 64   (bits 207:160)
    g1[6] = 0u;
    g1[7] = 0u;
    asm volatile("tensor_load_to_lds %0, %1" :: "s"(g0), "s"(g1) : "memory");
}

// ------------------------------------------------------------------
// Kernel 1: codebook -> bf16 hi/lo + 0.5*||e_hat||^2 (fp32)
// ------------------------------------------------------------------
__global__ void vq_prep_codebook(const float* __restrict__ cb,
                                 unsigned short* __restrict__ cb_hi,
                                 unsigned short* __restrict__ cb_lo,
                                 float* __restrict__ halfnorm) {
    int k = blockIdx.x * blockDim.x + threadIdx.x;
    if (k >= K_CODES) return;
    float s = 0.0f;
#pragma unroll 8
    for (int e = 0; e < E_DIM; ++e) {
        float v = cb[k * E_DIM + e];
        unsigned short h = f2bf_rne(v);
        float fh = bf2f(h);
        unsigned short l = f2bf_rne(v - fh);
        cb_hi[k * E_DIM + e] = h;
        cb_lo[k * E_DIM + e] = l;
        float r = fh + bf2f(l);
        s += r * r;
    }
    halfnorm[k] = 0.5f * s;
}

// ------------------------------------------------------------------
// Kernel 2: inputs [E,T,B] -> negated bf16 hi/lo, row-major [N, E]
// ------------------------------------------------------------------
__global__ __launch_bounds__(256)
void vq_prep_inputs(const float* __restrict__ in,
                    unsigned short* __restrict__ a_hi,
                    unsigned short* __restrict__ a_lo) {
    int n = blockIdx.x * blockDim.x + threadIdx.x;   // vector index (t*B + b)
    uint4* dh = (uint4*)(a_hi + (size_t)n * E_DIM);
    uint4* dl = (uint4*)(a_lo + (size_t)n * E_DIM);
#pragma unroll
    for (int g = 0; g < 8; ++g) {                    // 8 dims per 16B chunk
        uint4 h4, l4;
        unsigned short* hp = (unsigned short*)&h4;
        unsigned short* lp = (unsigned short*)&l4;
#pragma unroll
        for (int j = 0; j < 8; ++j) {
            int e = g * 8 + j;
            float v = -in[(size_t)e * N_VEC + n];    // negate: WMMA accumulates -x.e
            unsigned short h = f2bf_rne(v);
            hp[j] = h;
            lp[j] = f2bf_rne(v - bf2f(h));
        }
        dh[g] = h4;
        dl[g] = l4;
    }
}

// ------------------------------------------------------------------
// Kernel 3: fused GEMM + argmin with TDM double-buffered staging.
// Block = 256 threads = 8 waves; wave handles M=32 rows (2 WMMA row tiles).
// score = 0.5*||e||^2 - x.e  (3-term bf16 split GEMM, f32 accumulate)
// ------------------------------------------------------------------
__global__ __launch_bounds__(256)
void vq_argmin(const unsigned short* __restrict__ a_hi,
               const unsigned short* __restrict__ a_lo,
               const unsigned short* __restrict__ cb_hi,
               const unsigned short* __restrict__ cb_lo,
               const float* __restrict__ halfnorm,
               int* __restrict__ idx_out,
               float* __restrict__ counts) {
    // [buf][hi/lo][16 codes][64 dims padded to 72] : 144B rows (TDM pad), 2 buffers
    __shared__ __align__(16) unsigned short ldsB[2][2][16][72];
    __shared__ __align__(16) float ldsHN[K_CODES];

    const int tid  = threadIdx.x;
    const int lane = tid & 31;
    const int wave = tid >> 5;
    const int half = lane >> 4;    // 0: rows 0-7 / K-low chunks, 1: rows 8-15 / K-high
    const int l15  = lane & 15;
    const int waveRowBase = blockIdx.x * 256 + wave * 32;

    // kick off DMA of tile 0 into buffer 0 (hi + lo) before anything else
    if (wave == 0) {
        tdm_stage_tile(cb_hi, (unsigned)(size_t)&ldsB[0][0][0][0]);
        tdm_stage_tile(cb_lo, (unsigned)(size_t)&ldsB[0][1][0][0]);
    }

    // stage all 1024 halfnorm floats into LDS (one float4 per thread)
    ((float4*)ldsHN)[tid] = ((const float4*)halfnorm)[tid];

    union AV { uint4 q[2]; v16bf v; };

    // ---- load A operands once (reused across all 64 column tiles) ----
    // A in contiguous K order (same as B): lanes 0-15 hold K {0..15}; lanes 16-31 hold K {16..31}
    AV ah[2][2], al[2][2];
#pragma unroll
    for (int rt = 0; rt < 2; ++rt) {
        int row = waveRowBase + rt * 16 + l15;
        const unsigned short* ph = a_hi + (size_t)row * E_DIM;
        const unsigned short* pl = a_lo + (size_t)row * E_DIM;
#pragma unroll
        for (int ks = 0; ks < 2; ++ks) {
            int o = ks * 32 + half * 16;
            ah[rt][ks].q[0] = *(const uint4*)(ph + o);
            ah[rt][ks].q[1] = *(const uint4*)(ph + o + 8);
            al[rt][ks].q[0] = *(const uint4*)(pl + o);
            al[rt][ks].q[1] = *(const uint4*)(pl + o + 8);
        }
    }

    float best[2][8];
    int   bidx[2][8];
#pragma unroll
    for (int rt = 0; rt < 2; ++rt)
#pragma unroll
        for (int r = 0; r < 8; ++r) { best[rt][r] = 3.0e38f; bidx[rt][r] = 0; }

    for (int ct = 0; ct < N_TILES; ++ct) {
        const int buf = ct & 1;
        // wave 0: prefetch next tile into the other buffer, then wait for
        // tile ct's two DMAs (TDM ops complete in order per wave).
        if (wave == 0) {
            if (ct < N_TILES - 1) {
                const int nb = (ct + 1) & 1;
                tdm_stage_tile(cb_hi + (size_t)(ct + 1) * 16 * E_DIM,
                               (unsigned)(size_t)&ldsB[nb][0][0][0]);
                tdm_stage_tile(cb_lo + (size_t)(ct + 1) * 16 * E_DIM,
                               (unsigned)(size_t)&ldsB[nb][1][0][0]);
                __builtin_amdgcn_s_wait_tensorcnt(2);
            } else {
                __builtin_amdgcn_s_wait_tensorcnt(0);
            }
        }
        __syncthreads();   // tile ct visible to all waves (also covers ldsHN once)

        // B operands: contiguous K order (same as A): lane holds 16 contiguous K of its column
        AV bh[2], bl[2];
#pragma unroll
        for (int ks = 0; ks < 2; ++ks) {
            const unsigned short* ph = &ldsB[buf][0][l15][ks * 32 + half * 16];
            const unsigned short* pl = &ldsB[buf][1][l15][ks * 32 + half * 16];
            bh[ks].q[0] = *(const uint4*)(ph);
            bh[ks].q[1] = *(const uint4*)(ph + 8);
            bl[ks].q[0] = *(const uint4*)(pl);
            bl[ks].q[1] = *(const uint4*)(pl + 8);
        }

        float myhn = ldsHN[ct * 16 + l15];
        int   code = ct * 16 + l15;

#pragma unroll
        for (int rt = 0; rt < 2; ++rt) {
            v8f c = {0.f, 0.f, 0.f, 0.f, 0.f, 0.f, 0.f, 0.f};
#pragma unroll
            for (int ks = 0; ks < 2; ++ks) {
                c = __builtin_amdgcn_wmma_f32_16x16x32_bf16(
                        false, ah[rt][ks].v, false, bh[ks].v, (short)0, c, false, false);
                c = __builtin_amdgcn_wmma_f32_16x16x32_bf16(
                        false, ah[rt][ks].v, false, bl[ks].v, (short)0, c, false, false);
                c = __builtin_amdgcn_wmma_f32_16x16x32_bf16(
                        false, al[rt][ks].v, false, bh[ks].v, (short)0, c, false, false);
            }
#pragma unroll
            for (int r = 0; r < 8; ++r) {
                float s = c[r] + myhn;
                if (s < best[rt][r]) { best[rt][r] = s; bidx[rt][r] = code; }
            }
        }
        __syncthreads();   // all reads of buf done before it is DMA-overwritten
    }

    // cross-lane argmin within each 16-lane column group (tie -> lower index,
    // matching jnp.argmin first-minimum semantics)
#pragma unroll
    for (int rt = 0; rt < 2; ++rt) {
#pragma unroll
        for (int r = 0; r < 8; ++r) {
            float b = best[rt][r];
            int   i = bidx[rt][r];
#pragma unroll
            for (int m = 8; m >= 1; m >>= 1) {
                float ob = __shfl_xor(b, m, 32);
                int   oi = __shfl_xor(i, m, 32);
                if (ob < b || (ob == b && oi < i)) { b = ob; i = oi; }
            }
            if (l15 == 0) {
                int row = waveRowBase + rt * 16 + half * 8 + r;
                idx_out[row] = i;
                atomicAdd(&counts[i], 1.0f);
            }
        }
    }
}

// ------------------------------------------------------------------
// Kernel 4: gather + straight-through quantized + SSE reduction
// ------------------------------------------------------------------
__global__ __launch_bounds__(256)
void vq_gather(const float* __restrict__ in,
               const float* __restrict__ cb,
               const int* __restrict__ idx,
               float* __restrict__ out_q,
               float* __restrict__ sse) {
    int t = blockIdx.x * 256 + threadIdx.x;   // over E*N = 4194304
    int e = t >> 16;                           // / 65536
    int n = t & (N_VEC - 1);
    float x = in[t];
    float q = cb[idx[n] * E_DIM + e];
    out_q[t] = x + (q - x);                    // straight-through, as reference computes it
    float d = q - x;

    __shared__ float red[256];
    red[threadIdx.x] = d * d;
    __syncthreads();
#pragma unroll
    for (int s = 128; s > 0; s >>= 1) {
        if (threadIdx.x < s) red[threadIdx.x] += red[threadIdx.x + s];
        __syncthreads();
    }
    if (threadIdx.x == 0) atomicAdd(sse, red[0]);
}

// ------------------------------------------------------------------
// Kernel 5: finalize vq_loss + perplexity (single block)
// ------------------------------------------------------------------
__global__ __launch_bounds__(256)
void vq_finalize(const float* __restrict__ counts,
                 const float* __restrict__ sse,
                 float* __restrict__ out) {
    __shared__ float red[256];
    float s = 0.0f;
    for (int k = threadIdx.x; k < K_CODES; k += 256) {
        float p = counts[k] * (1.0f / (float)N_VEC);
        s += p * logf(p + 1e-10f);
    }
    red[threadIdx.x] = s;
    __syncthreads();
#pragma unroll
    for (int st = 128; st > 0; st >>= 1) {
        if (threadIdx.x < st) red[threadIdx.x] += red[threadIdx.x + st];
        __syncthreads();
    }
    if (threadIdx.x == 0) {
        float mse = sse[0] / (float)((size_t)N_VEC * E_DIM);
        out[0] = (1.0f + BETA) * mse;                       // vq_loss
        out[1 + (size_t)E_DIM * N_VEC] = expf(-red[0]);     // perplexity
    }
}

// ------------------------------------------------------------------
// Host launcher
// d_in[0] = inputs [E,T,B] f32 ; d_in[1] = codebook [K,E] f32
// d_out   = [vq_loss(1) | quantized(E*T*B) | perplexity(1)] f32
// ------------------------------------------------------------------
extern "C" void kernel_launch(void* const* d_in, const int* in_sizes, int n_in,
                              void* d_out, int out_size, void* d_ws, size_t ws_size,
                              hipStream_t stream) {
    const float* d_inputs   = (const float*)d_in[0];
    const float* d_codebook = (const float*)d_in[1];
    float* out = (float*)d_out;

    char* ws = (char*)d_ws;
    size_t off = 0;
    auto carve = [&](size_t bytes) -> void* {
        void* p = ws + off;
        off += (bytes + 255) & ~(size_t)255;
        return p;
    };
    unsigned short* a_hi  = (unsigned short*)carve((size_t)N_VEC * E_DIM * 2);   // 8 MB
    unsigned short* a_lo  = (unsigned short*)carve((size_t)N_VEC * E_DIM * 2);   // 8 MB
    unsigned short* cb_hi = (unsigned short*)carve((size_t)K_CODES * E_DIM * 2); // 128 KB
    unsigned short* cb_lo = (unsigned short*)carve((size_t)K_CODES * E_DIM * 2); // 128 KB
    float* halfnorm = (float*)carve(K_CODES * sizeof(float));
    int*   idx      = (int*)carve(N_VEC * sizeof(int));
    float* counts   = (float*)carve(K_CODES * sizeof(float));
    float* sse      = (float*)carve(256);

    hipMemsetAsync(counts, 0, K_CODES * sizeof(float), stream);
    hipMemsetAsync(sse, 0, sizeof(float), stream);

    vq_prep_codebook<<<(K_CODES + 255) / 256, 256, 0, stream>>>(d_codebook, cb_hi, cb_lo, halfnorm);
    vq_prep_inputs<<<N_VEC / 256, 256, 0, stream>>>(d_inputs, a_hi, a_lo);
    vq_argmin<<<N_VEC / 256, 256, 0, stream>>>(a_hi, a_lo, cb_hi, cb_lo, halfnorm, idx, counts);
    vq_gather<<<((size_t)N_VEC * E_DIM) / 256, 256, 0, stream>>>(d_inputs, d_codebook, idx, out + 1, sse);
    vq_finalize<<<1, 256, 0, stream>>>(counts, sse, out);
}
